// Decoder_86053964743383
// MI455X (gfx1250) — matrix operand order repaired and checked by reading, not compiled, not run on hardware
//
#include <hip/hip_runtime.h>
#include <hip/hip_bf16.h>

// Problem dims (fixed by the reference)
#define B_    32
#define S_    64
#define T_    64
#define E_    512
#define H_    1024
#define TWOE  1024
#define K3E   1536
#define KCAT  2560     // 3E (x) + H (h) fused GEMM depth
#define G4H   4096
#define V_    32000

typedef __bf16 v16bf __attribute__((ext_vector_type(16)));
typedef float  v8f   __attribute__((ext_vector_type(8)));

// ---------- CDNA5 async helpers ----------

#if __has_builtin(__builtin_amdgcn_s_wait_asynccnt)
#define WAIT_ASYNC(n) __builtin_amdgcn_s_wait_asynccnt(n)
#else
#define WAIT_ASYNC(n) asm volatile("s_wait_asynccnt %0" :: "n"(n) : "memory")
#endif

// Async copy 16 bytes global -> LDS (per lane). dst = 32-bit LDS byte address
// (low 32 bits of the generic pointer == addrspacecast to AS(3)).
__device__ __forceinline__ void async_copy_b128(const void* gsrc, void* lds_dst) {
    unsigned d = (unsigned)(unsigned long long)lds_dst;
    asm volatile("global_load_async_to_lds_b128 %0, %1, off"
                 :: "v"(d), "v"(gsrc) : "memory");
}

// ---------- math helpers ----------

__device__ __forceinline__ unsigned short f2bf(float f) {
    unsigned int u = __float_as_uint(f);
    unsigned int r = u + 0x7FFFu + ((u >> 16) & 1u);   // round-to-nearest-even
    return (unsigned short)(r >> 16);
}

__device__ __forceinline__ float sigmoidf_(float x) {
    return 1.0f / (1.0f + __expf(-x));
}

struct Frag { union { uint4 u[2]; v16bf v; }; };

// A fragment: 16x32 bf16 (MxK). ISA layout: lane L -> m = L&15, g = L>>4;
// VGPR0..3 hold K = g*8 .. g*8+7 ; VGPR4..7 hold K = 16 + g*8 .. (+7).
__device__ __forceinline__ v16bf load_a_frag(const unsigned short* __restrict__ A,
                                             int ld, int row0, int k0) {
    int lane = threadIdx.x & 31;
    int m = lane & 15, g = lane >> 4;
    const unsigned short* p = A + (size_t)(row0 + m) * ld + k0 + g * 8;
    Frag f;
    f.u[0] = *reinterpret_cast<const uint4*>(p);
    f.u[1] = *reinterpret_cast<const uint4*>(p + 16);
    return f.v;
}

// B fragment: 32x16 bf16 (KxN), sourced from row-major W[n][k] (k contiguous).
// Lane L -> n = L&15, khalf = L>>4; lane holds K = khalf*8 .. +7 and 16 + khalf*8 .. +7.
__device__ __forceinline__ v16bf load_b_frag(const unsigned short* __restrict__ W,
                                             int ld, int n0, int k0) {
    int lane = threadIdx.x & 31;
    int n = lane & 15, kh = lane >> 4;
    const unsigned short* p = W + (size_t)(n0 + n) * ld + k0 + kh * 8;
    Frag f;
    f.u[0] = *reinterpret_cast<const uint4*>(p);
    f.u[1] = *reinterpret_cast<const uint4*>(p + 16);
    return f.v;
}

__device__ __forceinline__ v8f wmma_bf16(v16bf a, v16bf b, v8f c) {
    return __builtin_amdgcn_wmma_f32_16x16x32_bf16(false, a, false, b,
                                                   (short)0, c, false, false);
}

// ---------- setup kernels ----------

__global__ void embed_kernel(const int* __restrict__ src, const int* __restrict__ pos,
                             const float* __restrict__ enc_emb,
                             const float* __restrict__ pos_emb,
                             float* __restrict__ enc) {
    int bs  = blockIdx.x;                       // 0 .. B*S-1
    int tok = src[bs];
    int pp  = pos[bs];
    float* dst = enc + (size_t)bs * TWOE;
    const float* we = enc_emb + (size_t)tok * E_;
    const float* pe = pos_emb + (size_t)pp * E_;
    for (int j = threadIdx.x; j < E_; j += blockDim.x) {
        dst[j]      = we[j];
        dst[E_ + j] = pe[j];
    }
}

__global__ void mean_kernel(const float* __restrict__ enc,
                            unsigned short* __restrict__ mean_bf) {
    int idx = blockIdx.x * blockDim.x + threadIdx.x;   // B*TWOE threads
    int b = idx >> 10, d = idx & 1023;
    const float* p = enc + (size_t)b * S_ * TWOE + d;
    float s = 0.f;
#pragma unroll 8
    for (int ss = 0; ss < S_; ++ss) s += p[ss * TWOE];
    mean_bf[idx] = f2bf(s * (1.0f / S_));
}

__global__ void cvt_bf16_kernel(const float* __restrict__ src,
                                unsigned short* __restrict__ dst, int n) {
    int i = blockIdx.x * blockDim.x + threadIdx.x;
    int stride = gridDim.x * blockDim.x;
    for (; i < n; i += stride) dst[i] = f2bf(src[i]);
}

__global__ void build_wcat_kernel(const float* __restrict__ Wih,
                                  const float* __restrict__ Whh,
                                  unsigned short* __restrict__ Wcat) {
    size_t i = (size_t)blockIdx.x * blockDim.x + threadIdx.x;
    size_t total = (size_t)G4H * KCAT;
    size_t stride = (size_t)gridDim.x * blockDim.x;
    for (; i < total; i += stride) {
        int n = (int)(i / KCAT), k = (int)(i % KCAT);
        float v = (k < K3E) ? Wih[(size_t)n * K3E + k]
                            : Whh[(size_t)n * H_ + (k - K3E)];
        Wcat[i] = f2bf(v);
    }
}

__global__ void bias_cat_kernel(const float* __restrict__ bih,
                                const float* __restrict__ bhh,
                                float* __restrict__ bc) {
    int i = blockIdx.x * blockDim.x + threadIdx.x;
    if (i < G4H) bc[i] = bih[i] + bhh[i];
}

// h0 = mean_enc @ W_h0^T + b_h0 ; c0 = h0.  M=32, N=1024, K=1024 (bf16 WMMA).
__global__ void h0_gemm_kernel(const unsigned short* __restrict__ meanA,
                               const unsigned short* __restrict__ Wh0,
                               const float* __restrict__ b_h0,
                               float* __restrict__ h, float* __restrict__ c) {
    int wave = threadIdx.x >> 5, lane = threadIdx.x & 31;
    int row0  = (wave & 1) * 16;
    int nbase = blockIdx.x * 128 + (wave >> 1) * 32;
    v8f acc0 = {}, acc1 = {};
    for (int k = 0; k < TWOE; k += 32) {
        v16bf a  = load_a_frag(meanA, TWOE, row0, k);
        v16bf b0 = load_b_frag(Wh0, TWOE, nbase,      k);
        v16bf b1 = load_b_frag(Wh0, TWOE, nbase + 16, k);
        acc0 = wmma_bf16(a, b0, acc0);
        acc1 = wmma_bf16(a, b1, acc1);
    }
    int nlo = lane & 15, grp = lane >> 4;
#pragma unroll
    for (int r = 0; r < 8; ++r) {
        int m  = row0 + r + 8 * grp;
        int n0 = nbase + nlo, n1 = nbase + 16 + nlo;
        float v0 = acc0[r] + b_h0[n0];
        float v1 = acc1[r] + b_h0[n1];
        h[m * H_ + n0] = v0; c[m * H_ + n0] = v0;
        h[m * H_ + n1] = v1; c[m * H_ + n1] = v1;
    }
}

// ---------- per-step kernels ----------

__global__ void attention_kernel(const float* __restrict__ enc,
                                 const float* __restrict__ h,
                                 const int* __restrict__ srclen,
                                 const int* __restrict__ tgt,
                                 const float* __restrict__ dec_emb,
                                 unsigned short* __restrict__ xh, int t) {
    __shared__ float part[64][4];
    __shared__ float e_s[64];
    __shared__ float a_s[64];
    int b   = blockIdx.x;
    int tid = threadIdx.x;
    const float* encb = enc + (size_t)b * S_ * TWOE;
    const float* hb   = h + b * H_;

    // energy[s] = dot(enc[b,s,:], h[b,:]) -- 4 threads per s
    int s = tid >> 2, q = tid & 3;
    {
        float sum = 0.f;
        const float* row = encb + s * TWOE + q * 256;
        const float* hh  = hb + q * 256;
#pragma unroll 8
        for (int k = 0; k < 256; ++k) sum += row[k] * hh[k];
        part[s][q] = sum;
    }
    __syncthreads();
    if (tid < 64) e_s[tid] = part[tid][0] + part[tid][1] + part[tid][2] + part[tid][3];
    __syncthreads();

    // masked softmax (softmax -> mask -> renormalize == exp*mask / sum(exp*mask))
    float mx = -1e30f;
    for (int i = 0; i < S_; ++i) mx = fmaxf(mx, e_s[i]);
    int L = srclen[b];
    if (tid < 64) a_s[tid] = (tid < L) ? __expf(e_s[tid] - mx) : 0.f;
    __syncthreads();
    float Z = 0.f;
    for (int i = 0; i < S_; ++i) Z += a_s[i];
    float rz = 1.0f / Z;

    // ctx[d] = sum_s a[s] * enc[b,s,d] ; 4 d's per thread
    unsigned short* xb = xh + b * KCAT;
    int d0 = tid * 4;
    float c0 = 0.f, c1 = 0.f, c2 = 0.f, c3 = 0.f;
    for (int ss = 0; ss < S_; ++ss) {
        float w = a_s[ss];
        const float* r4 = encb + ss * TWOE + d0;
        c0 += w * r4[0]; c1 += w * r4[1]; c2 += w * r4[2]; c3 += w * r4[3];
    }
    xb[E_ + d0 + 0] = f2bf(c0 * rz);
    xb[E_ + d0 + 1] = f2bf(c1 * rz);
    xb[E_ + d0 + 2] = f2bf(c2 * rz);
    xb[E_ + d0 + 3] = f2bf(c3 * rz);

    // word embedding (teacher forcing) and current h, in bf16
    int tok = tgt[b * T_ + t];
    const float* de = dec_emb + (size_t)tok * E_;
    for (int j = tid; j < E_; j += 256) xb[j] = f2bf(de[j]);
    for (int j = tid; j < H_; j += 256) xb[K3E + j] = f2bf(hb[j]);
}

// gates = xh @ Wcat^T + bias, then fused LSTM cell via LDS gate shuffle.
__global__ void gates_lstm_kernel(const unsigned short* __restrict__ xh,
                                  const unsigned short* __restrict__ Wcat,
                                  const float* __restrict__ bias,
                                  float* __restrict__ h, float* __restrict__ c,
                                  unsigned short* __restrict__ hs, int t) {
    __shared__ float gbuf[32][132];           // [batch][gate*32 + jj], padded
    int wave = threadIdx.x >> 5, lane = threadIdx.x & 31;
    int n0   = blockIdx.x * 32;               // cell-column chunk
    int row0 = (wave & 1) * 16;               // batch tile
    int g    = wave >> 1;                     // gate 0..3 (i,f,g,o)
    int nb   = g * H_ + n0;

    v8f acc0 = {}, acc1 = {};
    for (int k = 0; k < KCAT; k += 32) {
        v16bf a  = load_a_frag(xh, KCAT, row0, k);
        v16bf b0 = load_b_frag(Wcat, KCAT, nb,      k);
        v16bf b1 = load_b_frag(Wcat, KCAT, nb + 16, k);
        acc0 = wmma_bf16(a, b0, acc0);
        acc1 = wmma_bf16(a, b1, acc1);
        // stream-prefetch next K chunk of this wave's Wcat rows
        __builtin_prefetch(Wcat + (size_t)(nb + (lane & 15)) * KCAT + k + 64, 0, 0);
    }
    int nlo = lane & 15, grp = lane >> 4;
#pragma unroll
    for (int r = 0; r < 8; ++r) {
        int m = row0 + r + 8 * grp;
        gbuf[m][g * 32 + nlo]      = acc0[r];
        gbuf[m][g * 32 + 16 + nlo] = acc1[r];
    }
    __syncthreads();

#pragma unroll
    for (int it = 0; it < 4; ++it) {
        int idx = threadIdx.x + it * 256;     // 0..1023 -> (batch, jj)
        int b  = idx >> 5, jj = idx & 31;
        int n  = n0 + jj;
        float iG = gbuf[b][jj]      + bias[n];
        float fG = gbuf[b][32 + jj] + bias[H_ + n];
        float gG = gbuf[b][64 + jj] + bias[2 * H_ + n];
        float oG = gbuf[b][96 + jj] + bias[3 * H_ + n];
        float cn = sigmoidf_(fG) * c[b * H_ + n] + sigmoidf_(iG) * tanhf(gG);
        float hn = sigmoidf_(oG) * tanhf(cn);
        c[b * H_ + n] = cn;
        h[b * H_ + n] = hn;
        hs[((size_t)b * T_ + t) * H_ + n] = f2bf(hn);
    }
}

// ---------- output projection ----------
// out[b*T+t, v] = hs[b*T+t,:] . W_out[v,:] + b_out[v]
// WG tile 64M x 128N, K chunked by 64. A and B chunks are staged into LDS with
// async global->LDS copies (ASYNCcnt) and double-buffered against the WMMAs.
// LDS tile rows padded to 72 bf16 (144 B) for conflict-free ds_load_b128.
#define KC    64
#define LDP   72

__global__ __launch_bounds__(256)
void out_gemm_kernel(const unsigned short* __restrict__ hs,
                     const unsigned short* __restrict__ Wout,
                     const float* __restrict__ b_out,
                     float* __restrict__ out) {
    __shared__ unsigned short tA[2][64 * LDP];    // 2 x 9 KB
    __shared__ unsigned short tB[2][128 * LDP];   // 2 x 18 KB

    int tid   = threadIdx.x;
    int wave  = tid >> 5, lane = tid & 31;
    int mt    = wave & 3;                 // M sub-tile 0..3
    int nh    = wave >> 2;                // N half 0..1
    int mrow0 = blockIdx.y * 64;          // global M base of WG tile
    int nb0   = blockIdx.x * 128;         // global N base of WG tile

    // Stage one K-chunk (A: 64x64, B: 128x64 bf16) into LDS buffer `buf`.
    auto stage = [&](int buf, int kc0) {
#pragma unroll
        for (int j = 0; j < 2; ++j) {     // A: 512 lane-transfers of 16 B
            int idx = tid + j * 256;
            int row = idx >> 3, seg = idx & 7;
            const unsigned short* src = hs + (size_t)(mrow0 + row) * H_ + kc0 + seg * 8;
            async_copy_b128(src, &tA[buf][row * LDP + seg * 8]);
        }
#pragma unroll
        for (int j = 0; j < 4; ++j) {     // B: 1024 lane-transfers of 16 B
            int idx = tid + j * 256;
            int row = idx >> 3, seg = idx & 7;
            const unsigned short* src = Wout + (size_t)(nb0 + row) * H_ + kc0 + seg * 8;
            async_copy_b128(src, &tB[buf][row * LDP + seg * 8]);
        }
    };

    v8f acc[4] = {{}, {}, {}, {}};
    const int nch = H_ / KC;              // 16 chunks

    stage(0, 0);
    for (int ch = 0; ch < nch; ++ch) {
        if (ch + 1 < nch) {
            stage((ch + 1) & 1, (ch + 1) * KC);
            WAIT_ASYNC(6);                // 6 older asyncs (chunk ch) complete
        } else {
            WAIT_ASYNC(0);
        }
        __syncthreads();                  // chunk ch visible to all waves

        const unsigned short* Abuf = &tA[ch & 1][0];
        const unsigned short* Bbuf = &tB[ch & 1][0];
#pragma unroll
        for (int k2 = 0; k2 < KC; k2 += 32) {
            v16bf a = load_a_frag(Abuf, LDP, mt * 16, k2);
#pragma unroll
            for (int j = 0; j < 4; ++j) {
                v16bf b = load_b_frag(Bbuf, LDP, nh * 64 + j * 16, k2);
                acc[j] = wmma_bf16(a, b, acc[j]);
            }
        }
        __syncthreads();                  // all reads done before refill
    }

    int nlo = lane & 15, grp = lane >> 4;
#pragma unroll
    for (int j = 0; j < 4; ++j) {
        int n = nb0 + nh * 64 + j * 16 + nlo;
        float bb = b_out[n];
#pragma unroll
        for (int r = 0; r < 8; ++r) {
            int m = mrow0 + mt * 16 + r + 8 * grp;
            out[(size_t)m * V_ + n] = acc[j][r] + bb;
        }
    }
}

// ---------- launch ----------

extern "C" void kernel_launch(void* const* d_in, const int* in_sizes, int n_in,
                              void* d_out, int out_size, void* d_ws, size_t ws_size,
                              hipStream_t stream) {
    (void)in_sizes; (void)n_in; (void)out_size; (void)ws_size;
    const int*   tgt     = (const int*)d_in[0];
    const int*   srclen  = (const int*)d_in[2];
    const int*   src     = (const int*)d_in[3];
    const int*   pos     = (const int*)d_in[4];
    const float* enc_emb = (const float*)d_in[8];
    const float* pos_emb = (const float*)d_in[9];
    const float* dec_emb = (const float*)d_in[10];
    const float* W_h0    = (const float*)d_in[11];
    const float* b_h0    = (const float*)d_in[12];
    const float* W_ih    = (const float*)d_in[13];
    const float* W_hh    = (const float*)d_in[14];
    const float* b_ih    = (const float*)d_in[15];
    const float* b_hh    = (const float*)d_in[16];
    const float* W_out   = (const float*)d_in[17];
    const float* b_out   = (const float*)d_in[18];
    float* out = (float*)d_out;

    char* ws = (char*)d_ws;
    size_t off = 0;
    auto wsalloc = [&](size_t bytes) -> void* {
        void* p = ws + off;
        off = (off + bytes + 255) & ~(size_t)255;
        return p;
    };

    float*          enc    = (float*)         wsalloc((size_t)B_ * S_ * TWOE * 4); // 8 MB
    unsigned short* meanA  = (unsigned short*)wsalloc((size_t)B_ * TWOE * 2);
    float*          h      = (float*)         wsalloc((size_t)B_ * H_ * 4);
    float*          c      = (float*)         wsalloc((size_t)B_ * H_ * 4);
    unsigned short* xh     = (unsigned short*)wsalloc((size_t)B_ * KCAT * 2);
    unsigned short* hsbuf  = (unsigned short*)wsalloc((size_t)B_ * T_ * H_ * 2);   // 4 MB
    unsigned short* Wh0b   = (unsigned short*)wsalloc((size_t)H_ * TWOE * 2);      // 2 MB
    unsigned short* Wcatb  = (unsigned short*)wsalloc((size_t)G4H * KCAT * 2);     // 21 MB
    float*          biasc  = (float*)         wsalloc((size_t)G4H * 4);
    unsigned short* Woutb  = (unsigned short*)wsalloc((size_t)V_ * TWOE * 2);      // 64 MB

    // setup
    embed_kernel<<<B_ * S_, 256, 0, stream>>>(src, pos, enc_emb, pos_emb, enc);
    mean_kernel<<<(B_ * TWOE) / 256, 256, 0, stream>>>(enc, meanA);
    cvt_bf16_kernel<<<1024, 256, 0, stream>>>(W_h0, Wh0b, H_ * TWOE);
    build_wcat_kernel<<<2048, 256, 0, stream>>>(W_ih, W_hh, Wcatb);
    bias_cat_kernel<<<G4H / 256, 256, 0, stream>>>(b_ih, b_hh, biasc);
    cvt_bf16_kernel<<<4096, 256, 0, stream>>>(W_out, Woutb, V_ * TWOE);
    h0_gemm_kernel<<<H_ / 128, 256, 0, stream>>>(meanA, Wh0b, b_h0, h, c);

    // recurrence (graph-friendly: fixed 128 launches)
    for (int t = 0; t < T_; ++t) {
        attention_kernel<<<B_, 256, 0, stream>>>(enc, h, srclen, tgt, dec_emb, xh, t);
        gates_lstm_kernel<<<H_ / 32, 256, 0, stream>>>(xh, Wcatb, biasc, h, c, hsbuf, t);
    }

    // output projection
    dim3 og(V_ / 128, (B_ * T_) / 64);
    out_gemm_kernel<<<og, 256, 0, stream>>>(hsbuf, Woutb, b_out, out);
}
